// MLPAttention_52218212384987
// MI455X (gfx1250) — matrix operand order repaired and checked by reading, not compiled, not run on hardware
//
#include <hip/hip_runtime.h>
#include <hip/hip_bf16.h>
#include <math.h>

typedef __attribute__((ext_vector_type(16))) __bf16 v16bf;
typedef __attribute__((ext_vector_type(8)))  float   v8f;
typedef __attribute__((ext_vector_type(4)))  unsigned u32x4;
typedef __attribute__((ext_vector_type(8)))  unsigned u32x8;

#define B_  8
#define Q_  64
#define L_  1024
#define C_  512
#define D_  512
#define H_  256

// ---------------------------------------------------------------- fast tanh
__device__ __forceinline__ float fast_tanh(float x) {
#if __has_builtin(__builtin_amdgcn_tanhf)
  return __builtin_amdgcn_tanhf(x);            // v_tanh_f32
#else
  float r;
  asm("v_tanh_f32 %0, %1" : "=v"(r) : "v"(x));
  return r;
#endif
}

// ------------------------------------------------------- async copy helpers
__device__ __forceinline__ void async_copy_b128(const float* gsrc, const float* ldst) {
  unsigned l = (unsigned)(uintptr_t)ldst;      // low 32 bits = LDS byte offset
  asm volatile("global_load_async_to_lds_b128 %0, %1, off"
               :: "v"(l), "v"(gsrc) : "memory");
}
__device__ __forceinline__ void wait_async0() {
#if __has_builtin(__builtin_amdgcn_s_wait_asynccnt)
  __builtin_amdgcn_s_wait_asynccnt(0);
#else
  asm volatile("s_wait_asynccnt 0x0" ::: "memory");
#endif
}
__device__ __forceinline__ void wait_tensor0() {
#if __has_builtin(__builtin_amdgcn_s_wait_tensorcnt)
  __builtin_amdgcn_s_wait_tensorcnt(0);
#else
  asm volatile("s_wait_tensorcnt 0x0" ::: "memory");
#endif
}
// TDM: 1D contiguous block load with LDS padding (2 descriptor groups).
__device__ __forceinline__ void tdm_load_1d(unsigned lds_byte, unsigned long long gaddr,
                                            unsigned nelem_dw, unsigned dw0_g1) {
  u32x4 g0;
  g0[0] = 1u;                                   // count=1, user mode
  g0[1] = lds_byte;                             // lds_addr
  g0[2] = (unsigned)(gaddr & 0xffffffffull);    // global_addr[31:0]
  g0[3] = (unsigned)(gaddr >> 32) | (2u << 30); // global_addr[56:32] | type=2
  u32x8 g1;
  g1[0] = dw0_g1;                               // data_size | pad config
  g1[1] = (nelem_dw & 0xffffu) << 16;           // tensor_dim0[15:0]
  g1[2] = (nelem_dw >> 16) | (1u << 16);        // tensor_dim0[31:16] | tensor_dim1=1
  g1[3] = (nelem_dw & 0xffffu) << 16;           // tile_dim0[15:0]
  g1[4] = 1u;                                   // tile_dim1=1
  g1[5] = nelem_dw;                             // tensor_dim0_stride[31:0]
  g1[6] = 0u;
  g1[7] = 0u;
  asm volatile("tensor_load_to_lds %0, %1" :: "s"(g0), "s"(g1) : "memory");
}

// --------------------------------------------- WMMA fragment loaders (wave32)
// A 16x32 bf16: lane = row m=lane&15; K-groups {0-15}/{16-31}.
__device__ __forceinline__ v16bf load_a_frag(const float* At, int ldA, int mbase) {
  const int lane = threadIdx.x & 31;
  const float* row = At + (mbase + (lane & 15)) * ldA;
  const int kg = (lane >> 4) << 4;
  v16bf a;
#pragma unroll
  for (int i = 0; i < 4; ++i) {
    a[2*i]     = (__bf16)row[kg + 2*i];
    a[2*i + 1] = (__bf16)row[kg + 2*i + 1];
  }
#pragma unroll
  for (int i = 0; i < 4; ++i) {
    a[8 + 2*i]     = (__bf16)row[8 + kg + 2*i];
    a[8 + 2*i + 1] = (__bf16)row[8 + kg + 2*i + 1];
  }
  return a;
}
// B 32x16 bf16 from an n-major tile (Bt[n][k]): lane = col n=lane&15.
__device__ __forceinline__ v16bf load_b_fragN(const float* Bt, int ldB, int nbase) {
  const int lane = threadIdx.x & 31;
  const float* col = Bt + (nbase + (lane & 15)) * ldB;
  const int kg = (lane >> 4) << 4;
  v16bf b;
#pragma unroll
  for (int i = 0; i < 8; ++i) {
    b[2*i]     = (__bf16)col[kg + 2*i];
    b[2*i + 1] = (__bf16)col[kg + 2*i + 1];
  }
  return b;
}
// B 32x16 bf16 from a k-major tile (Bt[k][n]).
__device__ __forceinline__ v16bf load_b_fragK(const float* Bt, int ldB, int nbase) {
  const int lane = threadIdx.x & 31;
  const int n  = nbase + (lane & 15);
  const int kg = (lane >> 4) << 4;
  v16bf b;
#pragma unroll
  for (int i = 0; i < 8; ++i) {
    b[2*i]     = (__bf16)Bt[(kg + 2*i) * ldB + n];
    b[2*i + 1] = (__bf16)Bt[(kg + 2*i + 1) * ldB + n];
  }
  return b;
}

// ---------------------------------------------------------------- projection
// out[m,h] = sum_k A[m,k]*W[h,koff+k] (+bias). Block 128x64, waves 4x2 of 32x32.
// Async-to-LDS staging, double-buffered K loop.
__global__ __launch_bounds__(256) void proj_gemm_kernel(
    const float* __restrict__ A, int lda,
    const float* __restrict__ W, int ldw, int koff, int K,
    const float* __restrict__ bias,
    float* __restrict__ out, int Nld)
{
  __shared__ float At[2][128 * 36];
  __shared__ float Bt[2][64 * 36];   // n-major: Bt[n][k]
  const int t = threadIdx.x;
  const int hBase = blockIdx.x * 64;
  const int mBase = blockIdx.y * 128;
  const int w  = t >> 5;
  const int wm = (w >> 1) * 32;
  const int wn = (w & 1) * 32;

  v8f acc[2][2] = {};

  auto stage = [&](int k0, int buf) {
#pragma unroll
    for (int i = t; i < 128 * 8; i += 256) {      // A tile 128x32
      int r = i >> 3, c4 = (i & 7) * 4;
      async_copy_b128(&A[(size_t)(mBase + r) * lda + k0 + c4], &At[buf][r * 36 + c4]);
    }
#pragma unroll
    for (int i = t; i < 64 * 8; i += 256) {       // B tile: Bt[n][k] = W[h][koff+k]
      int n = i >> 3, c4 = (i & 7) * 4;
      async_copy_b128(&W[(size_t)(hBase + n) * ldw + koff + k0 + c4], &Bt[buf][n * 36 + c4]);
    }
  };

  stage(0, 0);
  const int NK = K / 32;
  for (int kk = 0; kk < NK; ++kk) {
    wait_async0();
    __syncthreads();                               // buffer kk published
    if (kk + 1 < NK) stage((kk + 1) * 32, (kk + 1) & 1);
    const float* Ab = At[kk & 1];
    const float* Bb = Bt[kk & 1];
    v16bf a0 = load_a_frag(Ab, 36, wm);
    v16bf a1 = load_a_frag(Ab, 36, wm + 16);
    v16bf b0 = load_b_fragN(Bb, 36, wn);
    v16bf b1 = load_b_fragN(Bb, 36, wn + 16);
    acc[0][0] = __builtin_amdgcn_wmma_f32_16x16x32_bf16(false, a0, false, b0, (short)0, acc[0][0], false, false);
    acc[0][1] = __builtin_amdgcn_wmma_f32_16x16x32_bf16(false, a0, false, b1, (short)0, acc[0][1], false, false);
    acc[1][0] = __builtin_amdgcn_wmma_f32_16x16x32_bf16(false, a1, false, b0, (short)0, acc[1][0], false, false);
    acc[1][1] = __builtin_amdgcn_wmma_f32_16x16x32_bf16(false, a1, false, b1, (short)0, acc[1][1], false, false);
  }

  const int lane = t & 31;
  const int nloc = lane & 15;
  const int mloc = (lane >> 4) * 8;
#pragma unroll
  for (int ti = 0; ti < 2; ++ti)
#pragma unroll
    for (int tj = 0; tj < 2; ++tj) {
      int gn = hBase + wn + tj * 16 + nloc;
      float bs = bias ? bias[gn] : 0.0f;
#pragma unroll
      for (int i = 0; i < 8; ++i) {
        int gm = mBase + wm + ti * 16 + mloc + i;
        out[(size_t)gm * Nld + gn] = acc[ti][tj][i] + bs;
      }
    }
}

// --------------------------------------------------- score + softmax (fused)
// Block = (b, 8 q). One wave owns one q; lane owns one l per 32-l chunk.
// ctx_proj chunks DMA'd by the Tensor Data Mover, double-buffered.
__global__ __launch_bounds__(256) void score_softmax_kernel(
    const float* __restrict__ ctx_proj,   // (B,L,H)
    const float* __restrict__ qb,         // (B,Q,H), bias included
    const float* __restrict__ v,
    const float* __restrict__ g,
    float* __restrict__ p)                // (B,Q,L)
{
  __shared__ float ctxS[2][32][260];      // 260 = 256 + TDM pad (4 dw / 256 dw)
  __shared__ float qbS[8][260];
  __shared__ float weff[H_];

  const int t   = threadIdx.x;
  const int b   = blockIdx.y;
  const int q0  = blockIdx.x * 8;
  const int q   = t >> 5;
  const int lof = t & 31;

  // w_eff = g * v / ||v||   (H == blockDim == 256)
  float vl = v[t];
  float* red = &ctxS[0][0][0];
  red[t] = vl * vl;
  __syncthreads();
  for (int s = 128; s > 0; s >>= 1) {
    if (t < s) red[t] += red[t + s];
    __syncthreads();
  }
  float nrm = sqrtf(red[0]);
  weff[t] = g[0] * vl / nrm;

  const float* qbg = qb + ((size_t)b * Q_ + q0) * H_;
#pragma unroll
  for (int i = t; i < 8 * 64; i += 256) {          // stage qb tile 8x256 (async)
    int r = i >> 6, c4 = (i & 63) * 4;
    async_copy_b128(&qbg[(size_t)r * H_ + c4], &qbS[r][c4]);
  }
  wait_async0();
  __syncthreads();   // red reads done, weff + qbS visible; ctxS now reusable

  const float* cpg = ctx_proj + (size_t)b * L_ * H_;
  // TDM group1 dw0: data_size=4B (2<<16) | pad_enable | pad_interval=256dw (7<<22)
  //                 | pad_amount=4dw (3<<25)
  const unsigned g1dw0 = (2u << 16) | (1u << 20) | (7u << 22) | (3u << 25);

  if (t < 32)
    tdm_load_1d((unsigned)(uintptr_t)&ctxS[0][0][0],
                (unsigned long long)(uintptr_t)cpg, 32 * H_, g1dw0);

  float scores[32];
  for (int ch = 0; ch < 32; ++ch) {
    if (t < 32) wait_tensor0();
    __syncthreads();                               // buffer ch&1 ready for all
    if (ch + 1 < 32 && t < 32)
      tdm_load_1d((unsigned)(uintptr_t)&ctxS[(ch + 1) & 1][0][0],
                  (unsigned long long)(uintptr_t)(cpg + (size_t)(ch + 1) * 32 * H_),
                  32 * H_, g1dw0);

    float s = 0.0f;
    const float4* cr = (const float4*)&ctxS[ch & 1][lof][0];
    const float4* qr = (const float4*)&qbS[q][0];
    const float4* wr = (const float4*)weff;
#pragma unroll 8
    for (int h4 = 0; h4 < H_ / 4; ++h4) {
      float4 cv = cr[h4], qv = qr[h4], wv = wr[h4];
      s = fmaf(wv.x, fast_tanh(cv.x + qv.x), s);
      s = fmaf(wv.y, fast_tanh(cv.y + qv.y), s);
      s = fmaf(wv.z, fast_tanh(cv.z + qv.z), s);
      s = fmaf(wv.w, fast_tanh(cv.w + qv.w), s);
    }
    scores[ch] = s;
  }

  // in-wave softmax over L=1024; sb is shift-invariant -> dropped.
  float mx = -3.402823466e38f;
#pragma unroll
  for (int ch = 0; ch < 32; ++ch) mx = fmaxf(mx, scores[ch]);
#pragma unroll
  for (int off = 16; off > 0; off >>= 1) mx = fmaxf(mx, __shfl_xor(mx, off, 32));
  float sum = 0.0f;
#pragma unroll
  for (int ch = 0; ch < 32; ++ch) {
    float e = __expf(scores[ch] - mx);
    scores[ch] = e;
    sum += e;
  }
#pragma unroll
  for (int off = 16; off > 0; off >>= 1) sum += __shfl_xor(sum, off, 32);
  const float inv = 1.0f / sum;
  float* pout = p + ((size_t)b * Q_ + (q0 + q)) * L_;
#pragma unroll
  for (int ch = 0; ch < 32; ++ch)
    pout[ch * 32 + lof] = scores[ch] * inv;
}

// ------------------------------------------------- expected = p @ context
// Per batch: (64x512) = (64x1024)(1024x512). Block 64x64, waves 4x2 of 16x32.
__global__ __launch_bounds__(256) void expected_ctx_kernel(
    const float* __restrict__ p,     // (B,Q,L)
    const float* __restrict__ ctx,   // (B,L,C)
    float* __restrict__ out)         // (B,Q,C)
{
  __shared__ float At[2][64 * 36];
  __shared__ float Bt[2][32 * 68];   // k-major: Bt[k][n]
  const int t = threadIdx.x;
  const int nBase = blockIdx.x * 64;
  const int b = blockIdx.y;
  const float* Ap = p   + (size_t)b * Q_ * L_;
  const float* Bp = ctx + (size_t)b * L_ * C_;

  const int w   = t >> 5;
  const int wmB = (w >> 1) * 16;
  const int wnB = (w & 1) * 32;

  v8f acc[2] = {};

  auto stage = [&](int k0, int buf) {
#pragma unroll
    for (int i = t; i < 64 * 8; i += 256) {       // A tile 64x32 (p)
      int r = i >> 3, c4 = (i & 7) * 4;
      async_copy_b128(&Ap[(size_t)r * L_ + k0 + c4], &At[buf][r * 36 + c4]);
    }
#pragma unroll
    for (int i = t; i < 32 * 16; i += 256) {      // B tile 32x64 (context)
      int k = i >> 4, c4 = (i & 15) * 4;
      async_copy_b128(&Bp[(size_t)(k0 + k) * C_ + nBase + c4], &Bt[buf][k * 68 + c4]);
    }
  };

  stage(0, 0);
  const int NK = L_ / 32;
  for (int kk = 0; kk < NK; ++kk) {
    wait_async0();
    __syncthreads();
    if (kk + 1 < NK) stage((kk + 1) * 32, (kk + 1) & 1);
    const float* Ab = At[kk & 1];
    const float* Bb = Bt[kk & 1];
    v16bf a  = load_a_frag(Ab, 36, wmB);
    v16bf b0 = load_b_fragK(Bb, 68, wnB);
    v16bf b1 = load_b_fragK(Bb, 68, wnB + 16);
    acc[0] = __builtin_amdgcn_wmma_f32_16x16x32_bf16(false, a, false, b0, (short)0, acc[0], false, false);
    acc[1] = __builtin_amdgcn_wmma_f32_16x16x32_bf16(false, a, false, b1, (short)0, acc[1], false, false);
  }

  const int lane = t & 31;
  const int nloc = lane & 15;
  const int mloc = (lane >> 4) * 8;
  float* outp = out + (size_t)b * Q_ * C_;
#pragma unroll
  for (int tj = 0; tj < 2; ++tj)
#pragma unroll
    for (int i = 0; i < 8; ++i) {
      int gm = wmB + mloc + i;                 // q
      int gn = nBase + wnB + tj * 16 + nloc;   // c
      outp[(size_t)gm * C_ + gn] = acc[tj][i];
    }
}

// ------------------------------------------------------------------- launch
extern "C" void kernel_launch(void* const* d_in, const int* in_sizes, int n_in,
                              void* d_out, int out_size, void* d_ws, size_t ws_size,
                              hipStream_t stream) {
  (void)in_sizes; (void)n_in; (void)out_size; (void)ws_size;
  const float* query   = (const float*)d_in[0];
  const float* context = (const float*)d_in[1];
  const float* W       = (const float*)d_in[2];
  const float* bias    = (const float*)d_in[3];
  const float* v       = (const float*)d_in[4];
  const float* g       = (const float*)d_in[5];
  // d_in[6] (sb): softmax shift-invariant -> no effect on outputs.

  float* ctx_proj = (float*)d_ws;                      // B*L*H fp32
  float* qb       = ctx_proj + (size_t)B_ * L_ * H_;   // B*Q*H fp32

  float* p_out  = (float*)d_out;                       // B*Q*L
  float* ec_out = p_out + (size_t)B_ * Q_ * L_;        // B*Q*C

  proj_gemm_kernel<<<dim3(H_ / 64, (B_ * L_) / 128), 256, 0, stream>>>(
      context, C_, W, C_ + D_, 0, C_, nullptr, ctx_proj, H_);
  proj_gemm_kernel<<<dim3(H_ / 64, (B_ * Q_) / 128), 256, 0, stream>>>(
      query, D_, W, C_ + D_, C_, D_, bias, qb, H_);
  score_softmax_kernel<<<dim3(Q_ / 8, B_), 256, 0, stream>>>(
      ctx_proj, qb, v, g, p_out);
  expected_ctx_kernel<<<dim3(C_ / 64, B_), 256, 0, stream>>>(
      p_out, context, ec_out);
}
